// Qwen3MoeSparseMoeBlock_55405078118962
// MI455X (gfx1250) — matrix operand order repaired and checked by reading, not compiled, not run on hardware
//
#include <hip/hip_runtime.h>

// Problem constants (Qwen3 MoE block)
#define T_TOK 2048
#define HID   2048
#define NEXP  64
#define IDIM  768
#define TOPK  8
#define MTILE 16   // tokens per block (WMMA N dimension)

typedef __attribute__((ext_vector_type(16))) __bf16 v16bf;
typedef __attribute__((ext_vector_type(8)))  float  v8f;

union Frag16 {
  v16bf v;
  uint4 q[2];
  unsigned int u[8];
};

// Hot-path fp32 pair -> packed bf16 by truncation: the high 16 bits of an
// fp32 are its truncated bf16, so one v_perm_b32 converts AND packs.
// result = { bf16(b) , bf16(a) }  (b in [31:16], a in [15:0])
__device__ __forceinline__ unsigned int pack2bf_trunc(float a, float b) {
  return __builtin_amdgcn_perm(__float_as_uint(b), __float_as_uint(a), 0x07060302u);
}
// Cold-path scalar convert with round-half-up bias (X staging, SwiGLU out).
__device__ __forceinline__ unsigned short f2bf(float f) {
  return (unsigned short)((__float_as_uint(f) + 0x8000u) >> 16);
}

// A-fragment (16x32 bf16): lane holds matrix row (lane&15); VGPR0-3 hold
// K = kg..kg+7, VGPR4-7 hold K = 16+kg..16+kg+7 with kg = (lane>>4)*8.
// p must point at row_base + k0 + kg (fp32 source, K-contiguous).
// 4x global_load_b128 + 8x v_perm_b32 per fragment.
__device__ __forceinline__ void load_a_frag(Frag16& a, const float* __restrict__ p) {
  float4 f0 = *(const float4*)(p);
  float4 f1 = *(const float4*)(p + 4);
  float4 f2 = *(const float4*)(p + 16);
  float4 f3 = *(const float4*)(p + 20);
  a.u[0] = pack2bf_trunc(f0.x, f0.y); a.u[1] = pack2bf_trunc(f0.z, f0.w);
  a.u[2] = pack2bf_trunc(f1.x, f1.y); a.u[3] = pack2bf_trunc(f1.z, f1.w);
  a.u[4] = pack2bf_trunc(f2.x, f2.y); a.u[5] = pack2bf_trunc(f2.z, f2.w);
  a.u[6] = pack2bf_trunc(f3.x, f3.y); a.u[7] = pack2bf_trunc(f3.z, f3.w);
}

// B-fragment (32x16 bf16): lane holds N column (lane&15); elements 0-7 hold
// K = kg..kg+7, 8-15 hold K = 16+kg..16+kg+7, kg = (lane>>4)*8.
// Source is k-major LDS tile (stride MTILE ushorts).
__device__ __forceinline__ void load_b_frag(Frag16& b, const unsigned short* __restrict__ lds,
                                            int k0, int lane) {
  const unsigned short* bp = lds + (size_t)(k0 + (lane >> 4) * 8) * MTILE + (lane & 15);
#pragma unroll
  for (int j = 0; j < 4; ++j) {
    b.u[j]     = (unsigned int)bp[(2 * j) * MTILE] | ((unsigned int)bp[(2 * j + 1) * MTILE] << 16);
    b.u[4 + j] = (unsigned int)bp[(16 + 2 * j) * MTILE] | ((unsigned int)bp[(17 + 2 * j) * MTILE] << 16);
  }
}

__device__ __forceinline__ v8f wmma_bf16(const Frag16& a, const Frag16& b, v8f c) {
  return __builtin_amdgcn_wmma_f32_16x16x32_bf16(false, a.v, false, b.v,
                                                 (short)0, c, false, false);
}

// fast SiLU: x * rcp(1 + exp(-x)) — v_exp_f32 + v_add + v_rcp + v_mul
__device__ __forceinline__ float fast_silu(float x) {
  return x * __builtin_amdgcn_rcpf(1.f + __expf(-x));
}

// ---------------------------------------------------------------------------
// Kernel 1: router.  One block per token (64 threads = one thread per expert).
// logits -> exp (softmax denominator cancels in top-k renorm) -> top-8 ->
// normalized weights -> atomic append into per-expert token lists.
// ---------------------------------------------------------------------------
__global__ void router_topk_kernel(const float* __restrict__ hs,
                                   const float* __restrict__ gw,
                                   int* __restrict__ cnt,
                                   int* __restrict__ toks,
                                   float* __restrict__ wts) {
  const int t = blockIdx.x;
  const int e = threadIdx.x;
  const float4* x4 = (const float4*)(hs + (size_t)t * HID);
  const float4* w4 = (const float4*)(gw + (size_t)e * HID);
  float acc = 0.f;
  for (int i = 0; i < HID / 4; ++i) {
    float4 a = x4[i], b = w4[i];
    acc += a.x * b.x + a.y * b.y + a.z * b.z + a.w * b.w;
  }
  __shared__ float lg[NEXP];
  lg[e] = acc;
  __syncthreads();
  if (e == 0) {
    float mx = lg[0];
    for (int i = 1; i < NEXP; ++i) mx = fmaxf(mx, lg[i]);
    float p[NEXP];
    for (int i = 0; i < NEXP; ++i) p[i] = __expf(lg[i] - mx);
    int   idx[TOPK];
    float val[TOPK];
    float s = 0.f;
    for (int k = 0; k < TOPK; ++k) {
      int bi = 0; float bv = -1.f;
      for (int i = 0; i < NEXP; ++i)
        if (p[i] > bv) { bv = p[i]; bi = i; }
      idx[k] = bi; val[k] = bv; p[bi] = -2.f; s += bv;
    }
    const float inv = 1.f / s;
    for (int k = 0; k < TOPK; ++k) {
      const int ex  = idx[k];
      const int pos = atomicAdd(&cnt[ex], 1);
      toks[(size_t)ex * T_TOK + pos] = t;
      wts [(size_t)ex * T_TOK + pos] = val[k] * inv;
    }
  }
}

// ---------------------------------------------------------------------------
// Kernel 2: grouped expert SwiGLU MLP.
// grid = NEXP * (T_TOK/MTILE) blocks (uniform early exit past the expert's
// real token count).  256 threads = 8 wave32.
// Phase A: D[i, tok] = sum_k W{g,u}[i,k] * X[tok,k]  (A = weight rows from
//          global fp32 -> bf16, B = X staged k-major in LDS), SwiGLU fused,
//          result stored k-major in LDS.
// Phase B: Y[h, tok] = sum_i Wd[h,i] * H[tok,i], scaled by routing weight,
//          scattered with f32 atomic adds into out.
// ---------------------------------------------------------------------------
__global__ void moe_expert_kernel(const float* __restrict__ hs,
                                  const float* __restrict__ wg_all,
                                  const float* __restrict__ wu_all,
                                  const float* __restrict__ wd_all,
                                  const int*   __restrict__ cnt,
                                  const int*   __restrict__ toks,
                                  const float* __restrict__ wts,
                                  float* __restrict__ out) {
  const int tilesPerE = T_TOK / MTILE;            // 128
  const int e    = blockIdx.x / tilesPerE;
  const int tile = blockIdx.x % tilesPerE;
  const int n_e  = cnt[e];
  const int base = tile * MTILE;
  if (base >= n_e) return;                        // uniform across block
  const int rows = (n_e - base < MTILE) ? (n_e - base) : MTILE;

  extern __shared__ unsigned short smem[];
  unsigned short* Xl = smem;                      // [HID][MTILE]  k-major, 64KB
  unsigned short* Hl = smem + (size_t)HID * MTILE;// [IDIM][MTILE] k-major, 24KB

  __shared__ int   tok[MTILE];
  __shared__ float twt[MTILE];

  const int tid    = threadIdx.x;
  const int lane   = tid & 31;
  const int wave   = tid >> 5;
  const int mrow   = lane & 15;                   // A-fragment row
  const int kgroup = (lane >> 4) * 8;             // A-fragment K sub-group

  if (tid < MTILE) {
    if (tid < rows) {
      tok[tid] = toks[(size_t)e * T_TOK + base + tid];
      twt[tid] = wts [(size_t)e * T_TOK + base + tid];
    } else {
      tok[tid] = 0;
      twt[tid] = 0.f;
    }
  }
  __syncthreads();

  // Stage X tile [MTILE tokens x HID] into LDS, transposed to k-major bf16.
  // Each thread converts 8 consecutive k of each row.
  for (int r = 0; r < MTILE; ++r) {
    float4 f0, f1;
    if (r < rows) {
      const float* src = hs + (size_t)tok[r] * HID + tid * 8;
      f0 = *(const float4*)(src);
      f1 = *(const float4*)(src + 4);
    } else {
      f0 = make_float4(0.f, 0.f, 0.f, 0.f);
      f1 = f0;
    }
    const int kb = tid * 8;
    Xl[(kb + 0) * MTILE + r] = f2bf(f0.x);
    Xl[(kb + 1) * MTILE + r] = f2bf(f0.y);
    Xl[(kb + 2) * MTILE + r] = f2bf(f0.z);
    Xl[(kb + 3) * MTILE + r] = f2bf(f0.w);
    Xl[(kb + 4) * MTILE + r] = f2bf(f1.x);
    Xl[(kb + 5) * MTILE + r] = f2bf(f1.y);
    Xl[(kb + 6) * MTILE + r] = f2bf(f1.z);
    Xl[(kb + 7) * MTILE + r] = f2bf(f1.w);
  }
  __syncthreads();

  // ---------------- Phase A: gate/up projections + SwiGLU -----------------
  const float* wg_e = wg_all + (size_t)e * IDIM * HID;
  const float* wu_e = wu_all + (size_t)e * IDIM * HID;

  for (int it = wave; it < IDIM / 16; it += 8) {
    const int i0 = it * 16;
    v8f cg = {0.f, 0.f, 0.f, 0.f, 0.f, 0.f, 0.f, 0.f};
    v8f cu = {0.f, 0.f, 0.f, 0.f, 0.f, 0.f, 0.f, 0.f};
    const float* gRow = wg_e + (size_t)(i0 + mrow) * HID;
    const float* uRow = wu_e + (size_t)(i0 + mrow) * HID;
#pragma unroll 2
    for (int k0 = 0; k0 < HID; k0 += 32) {
      Frag16 b, ag, au;
      load_b_frag(b, Xl, k0, lane);
      load_a_frag(ag, gRow + k0 + kgroup);
      load_a_frag(au, uRow + k0 + kgroup);
      cg = wmma_bf16(ag, b, cg);
      cu = wmma_bf16(au, b, cu);
    }
    // D layout: VGPR r -> row (lane>=16 ? 8+r : r), col = lane&15 (token)
    for (int r = 0; r < 8; ++r) {
      const int il   = (lane >> 4) * 8 + r;       // local intermediate index
      const int tcol = lane & 15;                 // token column
      const float h  = fast_silu(cg[r]) * cu[r];  // silu(g) * u
      Hl[(size_t)(i0 + il) * MTILE + tcol] = f2bf(h);
    }
  }
  __syncthreads();

  // ---------------- Phase B: down projection + weighted scatter -----------
  const float* wd_e = wd_all + (size_t)e * HID * IDIM;

  for (int ht = wave; ht < HID / 16; ht += 8) {
    const int h0 = ht * 16;
    v8f cy = {0.f, 0.f, 0.f, 0.f, 0.f, 0.f, 0.f, 0.f};
    const float* dRow = wd_e + (size_t)(h0 + mrow) * IDIM;
#pragma unroll 4
    for (int k0 = 0; k0 < IDIM; k0 += 32) {
      Frag16 b, ad;
      load_b_frag(b, Hl, k0, lane);
      load_a_frag(ad, dRow + k0 + kgroup);
      cy = wmma_bf16(ad, b, cy);
    }
    for (int r = 0; r < 8; ++r) {
      const int hl   = (lane >> 4) * 8 + r;       // local hidden index
      const int tcol = lane & 15;                 // token column
      const float v  = twt[tcol] * cy[r];
      unsafeAtomicAdd(&out[(size_t)tok[tcol] * HID + h0 + hl], v);
    }
  }
}

extern "C" void kernel_launch(void* const* d_in, const int* in_sizes, int n_in,
                              void* d_out, int out_size, void* d_ws, size_t ws_size,
                              hipStream_t stream) {
  (void)in_sizes; (void)n_in; (void)out_size; (void)ws_size;
  const float* hs = (const float*)d_in[0];   // [T, H]
  const float* gw = (const float*)d_in[1];   // [E, H]
  const float* wg = (const float*)d_in[2];   // [E, I, H]
  const float* wu = (const float*)d_in[3];   // [E, I, H]
  const float* wd = (const float*)d_in[4];   // [E, H, I]
  float* out = (float*)d_out;                // [T, H]

  // Workspace layout: counts | token lists | weight lists  (~1 MB)
  int*   cnt  = (int*)d_ws;
  int*   toks = (int*)((char*)d_ws + 256);
  float* wts  = (float*)((char*)d_ws + 256 + (size_t)NEXP * T_TOK * sizeof(int));

  hipMemsetAsync(cnt, 0, 256, stream);
  hipMemsetAsync(out, 0, (size_t)T_TOK * HID * sizeof(float), stream);

  router_topk_kernel<<<T_TOK, NEXP, 0, stream>>>(hs, gw, cnt, toks, wts);

  const dim3 grid(NEXP * (T_TOK / MTILE));       // 8192 blocks, early-exit
  const size_t smem =
      ((size_t)HID * MTILE + (size_t)IDIM * MTILE) * sizeof(unsigned short); // 88KB
  moe_expert_kernel<<<grid, 256, smem, stream>>>(hs, wg, wu, wd, cnt, toks, wts, out);
}
